// Pointnet2MSG_5360119186136
// MI455X (gfx1250) — matrix operand order repaired and checked by reading, not compiled, not run on hardware
//
#include <hip/hip_runtime.h>
#include <math.h>

// ---------------------------------------------------------------------------
// CDNA5 (gfx1250) wave32 WMMA / TDM types
// ---------------------------------------------------------------------------
typedef __attribute__((ext_vector_type(16))) _Float16     v16h;
typedef __attribute__((ext_vector_type(4)))  _Float16     v4h;
typedef __attribute__((ext_vector_type(8)))  float        v8f;
typedef __attribute__((ext_vector_type(4)))  unsigned int v4u;
typedef __attribute__((ext_vector_type(8)))  int          v8i;
typedef __attribute__((ext_vector_type(4)))  int          v4i;

// ---------------------------------------------------------------------------
// TDM: async-DMA a 32-row x 32-col f32 tile (row stride K) from global into
// LDS, with OOB zero-fill via tensor_dim0/1 = remaining K/O. 2-D descriptor:
//   group0: count=1 | lds_addr | global_addr[56:0] | type=2
//   group1: data_size=4B | tensor_dim0=remK | tensor_dim1=remO |
//           tile_dim0=32 | tile_dim1=32 | tensor_dim0_stride=K
// Tracked by TENSORcnt (s_wait_tensorcnt).
// ---------------------------------------------------------------------------
__device__ __forceinline__ void tdm_load_tile_a(const float* gsrc, unsigned lds_off,
                                                int remK, int remO, int K)
{
    unsigned long long ga = (unsigned long long)(uintptr_t)gsrc;
    v4u g0;
    g0[0] = 1u;                                             // count=1 (user D#)
    g0[1] = lds_off;                                        // lds_addr (bytes)
    g0[2] = (unsigned)(ga & 0xFFFFFFFFu);                   // global_addr[31:0]
    g0[3] = (unsigned)((ga >> 32) & 0x01FFFFFFu) | (2u << 30); // addr[56:32]|type=2
    v8i g1;
    g1[0] = (int)(2u << 16);                                // data_size: 2 => 4 bytes
    g1[1] = (int)(((unsigned)remK & 0xFFFFu) << 16);        // tensor_dim0[15:0]
    g1[2] = (int)((((unsigned)remK >> 16) & 0xFFFFu) |
                  (((unsigned)remO & 0xFFFFu) << 16));      // dim0[31:16]|dim1[15:0]
    g1[3] = (int)((((unsigned)remO >> 16) & 0xFFFFu) |
                  (32u << 16));                             // dim1[31:16]|tile_dim0=32
    g1[4] = 32;                                             // tile_dim1=32, tile_dim2=0
    g1[5] = K;                                              // tensor_dim0_stride[31:0]
    g1[6] = 0;                                              // stride[47:32]|dim1_stride
    g1[7] = 0;
    v4i z4 = {};
#if __clang_major__ >= 23
    v8i z8 = {};
    __builtin_amdgcn_tensor_load_to_lds(g0, g1, z4, z4, z8, 0);
#else
    __builtin_amdgcn_tensor_load_to_lds(g0, g1, z4, z4, 0);
#endif
}

// ---------------------------------------------------------------------------
// Fused GEMM  Y[OxN] = W[OxK] * X (+bias), f16 WMMA, f32 accumulate.
// mode 0: X dense [K x N] row-major.
// mode 1: implicit im2col conv3x3 (pad 1, stride s): K = Cin*9, N = Hout*Wout,
//         X = input feature map [Cin x Hin x Win].
// Block = 128 threads = 4 wave32s. Block tile = 32 (O) x 128 (N); K-step 32.
// A tile: TDM async global->LDS (f32), then LDS->LDS f16 convert into WMMA
// fragment order (2x ds_load_b128 per fragment read). Each wave gathers two
// per-lane B fragments from global (coalesced dword loads) and issues 4 WMMAs
// per K-step (a0/a1 x b0/b1). Fast path: zero guards; general path:
// clamp+select only (TDM zero-fills A out-of-bounds in both).
// ---------------------------------------------------------------------------
__global__ void __launch_bounds__(128) wmma_gemm_kernel(
    const float* __restrict__ Wm, const float* __restrict__ X,
    const float* __restrict__ bias, float* __restrict__ Y,
    int O, int K, int N, int mode,
    int Hin, int Win, int Hout, int Wout, int stride)
{
    __shared__ __align__(16) float    sA32[32 * 32];       // TDM destination
    __shared__ __align__(16) _Float16 sAf[2 * 32 * 16];    // 2 O-tiles, frag order

    const int tid   = threadIdx.x;
    const int wav   = tid >> 5;       // 0..3
    const int lane  = tid & 31;
    const int hf    = lane >> 4;      // lane half
    const int l16   = lane & 15;
    const int oBase = blockIdx.y * 32;
    const int nBase = blockIdx.x * 128;
    const unsigned ldsA = (unsigned)(uintptr_t)(void*)sA32;

    const int  n0  = nBase + wav * 16 + l16;
    const int  n1  = n0 + 64;
    const bool ok0 = (n0 < N), ok1 = (n1 < N);
    const int  nc0 = ok0 ? n0 : (N - 1);
    const int  nc1 = ok1 ? n1 : (N - 1);
    int oh0 = 0, ow0 = 0, oh1 = 0, ow1 = 0;
    if (mode != 0) {
        oh0 = nc0 / Wout; ow0 = nc0 % Wout;
        oh1 = nc1 / Wout; ow1 = nc1 % Wout;
    }

    v8f acc00 = {}, acc01 = {}, acc10 = {}, acc11 = {};

    const bool fast = (oBase + 32 <= O) && (nBase + 128 <= N) && ((K & 31) == 0);

    if (fast) {
        for (int k0 = 0; k0 < K; k0 += 32) {
            // ---- A tile: TDM global->LDS, then f16 convert to frag order ----
            tdm_load_tile_a(Wm + (size_t)oBase * K + k0, ldsA, K - k0, O - oBase, K);
            __builtin_amdgcn_s_wait_tensorcnt(0);
            __syncthreads();
#pragma unroll
            for (int e = 0; e < 2; ++e) {
                int r    = tid + 128 * e;
                int tile = r >> 7;
                int L    = (r >> 2) & 31;
                int blk  = r & 3;
                int m    = tile * 16 + (L & 15);
                int kk   = 16 * (L >> 4) + 4 * blk;
                const float* p = sA32 + m * 32 + kk;
                v4h h4;
                h4[0] = (_Float16)p[0]; h4[1] = (_Float16)p[1];
                h4[2] = (_Float16)p[2]; h4[3] = (_Float16)p[3];
                *(v4h*)(sAf + (size_t)(tile * 32 + L) * 16 + 4 * blk) = h4;
            }
            __syncthreads();

            v16h a0 = *(const v16h*)(sAf + (size_t)lane * 16);
            v16h a1 = *(const v16h*)(sAf + (size_t)(32 + lane) * 16);

            v16h b0, b1;
            if (mode == 0) {
#pragma unroll
                for (int h = 0; h < 16; ++h) {
                    size_t row = (size_t)(k0 + hf * 16 + h) * N;
                    b0[h] = (_Float16)X[row + n0];
                    b1[h] = (_Float16)X[row + n1];
                }
                if (k0 + 32 < K)
                    __builtin_prefetch(&X[(size_t)(k0 + 32) * N + n0], 0, 1);
            } else {
#pragma unroll
                for (int h = 0; h < 16; ++h) {
                    int k = k0 + hf * 16 + h;
                    int c = k / 9, rs = k % 9;
                    int dr = rs / 3 - 1, ds = rs % 3 - 1;
                    int ih0 = oh0 * stride + dr, iw0 = ow0 * stride + ds;
                    int ih1 = oh1 * stride + dr, iw1 = ow1 * stride + ds;
                    bool in0 = ih0 >= 0 && ih0 < Hin && iw0 >= 0 && iw0 < Win;
                    bool in1 = ih1 >= 0 && ih1 < Hin && iw1 >= 0 && iw1 < Win;
                    float v0 = X[((size_t)c * Hin + (in0 ? ih0 : 0)) * Win + (in0 ? iw0 : 0)];
                    float v1 = X[((size_t)c * Hin + (in1 ? ih1 : 0)) * Win + (in1 ? iw1 : 0)];
                    b0[h] = (_Float16)(in0 ? v0 : 0.f);
                    b1[h] = (_Float16)(in1 ? v1 : 0.f);
                }
            }

            acc00 = __builtin_amdgcn_wmma_f32_16x16x32_f16(false, a0, false, b0, (short)0, acc00, false, false);
            acc10 = __builtin_amdgcn_wmma_f32_16x16x32_f16(false, a1, false, b0, (short)0, acc10, false, false);
            acc01 = __builtin_amdgcn_wmma_f32_16x16x32_f16(false, a0, false, b1, (short)0, acc01, false, false);
            acc11 = __builtin_amdgcn_wmma_f32_16x16x32_f16(false, a1, false, b1, (short)0, acc11, false, false);
            __syncthreads();
        }
    } else {
        for (int k0 = 0; k0 < K; k0 += 32) {
            // ---- A tile via TDM (OOB rows/cols zero-filled by the DMA) -----
            tdm_load_tile_a(Wm + (size_t)oBase * K + k0, ldsA, K - k0, O - oBase, K);
            __builtin_amdgcn_s_wait_tensorcnt(0);
            __syncthreads();
#pragma unroll
            for (int e = 0; e < 2; ++e) {
                int r    = tid + 128 * e;
                int tile = r >> 7;
                int L    = (r >> 2) & 31;
                int blk  = r & 3;
                int m    = tile * 16 + (L & 15);
                int kk   = 16 * (L >> 4) + 4 * blk;
                const float* p = sA32 + m * 32 + kk;
                v4h h4;
                h4[0] = (_Float16)p[0]; h4[1] = (_Float16)p[1];
                h4[2] = (_Float16)p[2]; h4[3] = (_Float16)p[3];
                *(v4h*)(sAf + (size_t)(tile * 32 + L) * 16 + 4 * blk) = h4;
            }
            __syncthreads();

            v16h a0 = *(const v16h*)(sAf + (size_t)lane * 16);
            v16h a1 = *(const v16h*)(sAf + (size_t)(32 + lane) * 16);

            v16h b0, b1;
            if (mode == 0) {
#pragma unroll
                for (int h = 0; h < 16; ++h) {
                    int k   = k0 + hf * 16 + h;
                    int kcl = (k < K) ? k : (K - 1);
                    size_t row = (size_t)kcl * N;
                    float v0 = X[row + nc0];
                    float v1 = X[row + nc1];
                    b0[h] = (_Float16)((k < K && ok0) ? v0 : 0.f);
                    b1[h] = (_Float16)((k < K && ok1) ? v1 : 0.f);
                }
            } else {
#pragma unroll
                for (int h = 0; h < 16; ++h) {
                    int k   = k0 + hf * 16 + h;
                    int kcl = (k < K) ? k : 0;
                    int c = kcl / 9, rs = kcl % 9;
                    int dr = rs / 3 - 1, ds = rs % 3 - 1;
                    int ih0 = oh0 * stride + dr, iw0 = ow0 * stride + ds;
                    int ih1 = oh1 * stride + dr, iw1 = ow1 * stride + ds;
                    bool in0 = (k < K) && ok0 && ih0 >= 0 && ih0 < Hin && iw0 >= 0 && iw0 < Win;
                    bool in1 = (k < K) && ok1 && ih1 >= 0 && ih1 < Hin && iw1 >= 0 && iw1 < Win;
                    float v0 = X[((size_t)c * Hin + (in0 ? ih0 : 0)) * Win + (in0 ? iw0 : 0)];
                    float v1 = X[((size_t)c * Hin + (in1 ? ih1 : 0)) * Win + (in1 ? iw1 : 0)];
                    b0[h] = (_Float16)(in0 ? v0 : 0.f);
                    b1[h] = (_Float16)(in1 ? v1 : 0.f);
                }
            }

            acc00 = __builtin_amdgcn_wmma_f32_16x16x32_f16(false, a0, false, b0, (short)0, acc00, false, false);
            acc10 = __builtin_amdgcn_wmma_f32_16x16x32_f16(false, a1, false, b0, (short)0, acc10, false, false);
            acc01 = __builtin_amdgcn_wmma_f32_16x16x32_f16(false, a0, false, b1, (short)0, acc01, false, false);
            acc11 = __builtin_amdgcn_wmma_f32_16x16x32_f16(false, a1, false, b1, (short)0, acc11, false, false);
            __syncthreads();
        }
    }

    // -------- store D tiles (C/D layout: M = v + 8*hf, N = l16) -------------
#pragma unroll
    for (int v = 0; v < 8; ++v) {
        int o0 = oBase + v + 8 * hf;
        int o1 = o0 + 16;
        float b0v = 0.f, b1v = 0.f;
        if (bias) {
            b0v = (o0 < O) ? bias[o0] : 0.f;
            b1v = (o1 < O) ? bias[o1] : 0.f;
        }
        if (ok0 && o0 < O) Y[(size_t)o0 * N + n0] = acc00[v] + b0v;
        if (ok0 && o1 < O) Y[(size_t)o1 * N + n0] = acc10[v] + b1v;
        if (ok1 && o0 < O) Y[(size_t)o0 * N + n1] = acc01[v] + b0v;
        if (ok1 && o1 < O) Y[(size_t)o1 * N + n1] = acc11[v] + b1v;
    }
}

// ---------------------------------------------------------------------------
// BatchNorm (batch-stat) helpers
// ---------------------------------------------------------------------------
__global__ void stats_kernel(const float* __restrict__ Y, int N, float* __restrict__ mv)
{
    int c = blockIdx.x;
    const float* row = Y + (size_t)c * N;
    __shared__ float ss[256], sq[256];
    float s = 0.f, q = 0.f;
    for (int i = threadIdx.x; i < N; i += 256) { float v = row[i]; s += v; q += v * v; }
    ss[threadIdx.x] = s; sq[threadIdx.x] = q;
    __syncthreads();
    for (int st = 128; st > 0; st >>= 1) {
        if (threadIdx.x < st) { ss[threadIdx.x] += ss[threadIdx.x + st]; sq[threadIdx.x] += sq[threadIdx.x + st]; }
        __syncthreads();
    }
    if (threadIdx.x == 0) {
        float m = ss[0] / (float)N;
        mv[2 * c] = m;
        mv[2 * c + 1] = sq[0] / (float)N - m * m;
    }
}

__global__ void bn_relu_kernel(float* __restrict__ Y, const float* __restrict__ mv,
                               const float* __restrict__ g, const float* __restrict__ b,
                               int O, int N, int relu)
{
    long idx = (long)blockIdx.x * blockDim.x + threadIdx.x;
    if (idx >= (long)O * N) return;
    int c = (int)(idx / N);
    float y = (Y[idx] - mv[2 * c]) * rsqrtf(mv[2 * c + 1] + 1e-5f) * g[c] + b[c];
    Y[idx] = relu ? fmaxf(y, 0.f) : y;
}

// ---------------------------------------------------------------------------
// Farthest point sampling (single persistent block)
// ---------------------------------------------------------------------------
__global__ void fps_kernel(const float* __restrict__ xyz, int N, int npoint,
                           int* __restrict__ out, float* __restrict__ dist)
{
    const int t = threadIdx.x;
    for (int i = t; i < N; i += blockDim.x) dist[i] = 1e10f;
    __shared__ float sval[1024];
    __shared__ int   sidx[1024];
    __shared__ int   cur;
    if (t == 0) cur = 0;
    __syncthreads();
    for (int it = 0; it < npoint; ++it) {
        if (t == 0) out[it] = cur;
        __syncthreads();
        float cx = xyz[3 * cur], cy = xyz[3 * cur + 1], cz = xyz[3 * cur + 2];
        float best = -1.f; int bi = 0;
        for (int i = t; i < N; i += blockDim.x) {
            float dx = xyz[3 * i] - cx, dy = xyz[3 * i + 1] - cy, dz = xyz[3 * i + 2] - cz;
            float d = dx * dx + dy * dy + dz * dz;
            float nd = fminf(dist[i], d);
            dist[i] = nd;
            if (nd > best) { best = nd; bi = i; }
        }
        sval[t] = best; sidx[t] = bi;
        __syncthreads();
        for (int s = blockDim.x >> 1; s > 0; s >>= 1) {
            if (t < s && sval[t + s] > sval[t]) { sval[t] = sval[t + s]; sidx[t] = sidx[t + s]; }
            __syncthreads();
        }
        if (t == 0) cur = sidx[0];
        __syncthreads();
    }
}

// ---------------------------------------------------------------------------
// Gathers / grouping
// ---------------------------------------------------------------------------
__global__ void gather3_kernel(const float* src, const int* idx, float* dst, int n)
{
    int i = blockIdx.x * blockDim.x + threadIdx.x;
    if (i >= n) return;
    int j = idx[i];
    dst[3 * i] = src[3 * j]; dst[3 * i + 1] = src[3 * j + 1]; dst[3 * i + 2] = src[3 * j + 2];
}

__global__ void gather2_kernel(const float* src, const int* idx, float* dst, int n)
{
    int i = blockIdx.x * blockDim.x + threadIdx.x;
    if (i >= n) return;
    int j = idx[i];
    dst[2 * i] = src[2 * j]; dst[2 * i + 1] = src[2 * j + 1];
}

__global__ void ballq_kernel(const float* __restrict__ nxyz, const float* __restrict__ xyz,
                             int np, int N, float r2, int ns, int* __restrict__ gidx)
{
    int i = blockIdx.x * blockDim.x + threadIdx.x;
    if (i >= np) return;
    float cx = nxyz[3 * i], cy = nxyz[3 * i + 1], cz = nxyz[3 * i + 2];
    int cnt = 0, first = 0; bool has = false;
    for (int j = 0; j < N && cnt < ns; ++j) {
        float dx = xyz[3 * j] - cx, dy = xyz[3 * j + 1] - cy, dz = xyz[3 * j + 2] - cz;
        if (dx * dx + dy * dy + dz * dz < r2) {
            if (!has) { first = j; has = true; }
            gidx[(size_t)i * ns + cnt] = j; ++cnt;
        }
    }
    int pad = has ? first : 0;
    for (; cnt < ns; ++cnt) gidx[(size_t)i * ns + cnt] = pad;
}

__global__ void build_group_kernel(const float* __restrict__ xyz, const float* __restrict__ feat,
                                   int Cin, int N, const int* __restrict__ gidx,
                                   const float* __restrict__ nxyz, int np, int ns,
                                   float* __restrict__ X)
{
    int idx = blockIdx.x * blockDim.x + threadIdx.x;
    int tot = np * ns;
    if (idx >= tot) return;
    int i = idx / ns;
    int j = gidx[idx];
    size_t NN = (size_t)tot;
    X[0 * NN + idx] = xyz[3 * j]     - nxyz[3 * i];
    X[1 * NN + idx] = xyz[3 * j + 1] - nxyz[3 * i + 1];
    X[2 * NN + idx] = xyz[3 * j + 2] - nxyz[3 * i + 2];
    for (int c = 0; c < Cin; ++c)
        X[(size_t)(3 + c) * NN + idx] = feat[(size_t)c * N + j];
}

__global__ void rowmax_kernel(const float* __restrict__ X, int C, int np, int ns,
                              float* __restrict__ out)
{
    int idx = blockIdx.x * blockDim.x + threadIdx.x;
    if (idx >= C * np) return;
    int c = idx / np, i = idx % np;
    const float* p = X + (size_t)c * np * ns + (size_t)i * ns;
    float m = p[0];
    for (int s = 1; s < ns; ++s) m = fmaxf(m, p[s]);
    out[(size_t)c * np + i] = m;
}

// ---------------------------------------------------------------------------
// Bilinear grid sample (zero-pad), normalized grid in [-1,1]
// ---------------------------------------------------------------------------
__global__ void grid_sample_kernel(const float* __restrict__ fm, int C, int H, int W,
                                   const float* __restrict__ grid, int n,
                                   float* __restrict__ out)
{
    int idx = blockIdx.x * blockDim.x + threadIdx.x;
    if (idx >= C * n) return;
    int c = idx / n, i = idx % n;
    float gx = (grid[2 * i] + 1.f) * (W * 0.5f) - 0.5f;
    float gy = (grid[2 * i + 1] + 1.f) * (H * 0.5f) - 0.5f;
    int x0 = (int)floorf(gx), y0 = (int)floorf(gy);
    float acc = 0.f;
    for (int dy = 0; dy < 2; ++dy)
        for (int dx = 0; dx < 2; ++dx) {
            int xx = x0 + dx, yy = y0 + dy;
            float wx = dx ? (gx - (float)x0) : ((float)x0 + 1.f - gx);
            float wy = dy ? (gy - (float)y0) : ((float)y0 + 1.f - gy);
            if (xx >= 0 && xx < W && yy >= 0 && yy < H)
                acc += fm[((size_t)c * H + yy) * W + xx] * wx * wy;
        }
    out[(size_t)c * n + i] = acc;
}

// ---------------------------------------------------------------------------
// Misc elementwise
// ---------------------------------------------------------------------------
__global__ void xn_kernel(const float* xy, float* xn, int n)
{
    int i = blockIdx.x * blockDim.x + threadIdx.x;
    if (i >= n) return;
    xn[2 * i]     = xy[2 * i]     / 1279.f * 2.f - 1.f;
    xn[2 * i + 1] = xy[2 * i + 1] / 383.f  * 2.f - 1.f;
}

__global__ void add_tanh_kernel(float* a, const float* b, int n)
{
    int i = blockIdx.x * blockDim.x + threadIdx.x;
    if (i < n) a[i] = tanhf(a[i] + b[i]);
}

__global__ void sigmoid_kernel(float* a, int n)
{
    int i = blockIdx.x * blockDim.x + threadIdx.x;
    if (i < n) a[i] = 1.f / (1.f + expf(-a[i]));
}

__global__ void mul_att_kernel(float* Y, const float* att, int C, int n)
{
    int idx = blockIdx.x * blockDim.x + threadIdx.x;
    if (idx >= C * n) return;
    Y[idx] *= att[idx % n];
}

// ---------------------------------------------------------------------------
// FP: 3-NN inverse-squared-distance interpolation
// ---------------------------------------------------------------------------
__global__ void fp_interp_kernel(const float* __restrict__ ux, const float* __restrict__ kx,
                                 const float* __restrict__ kf, int C, int nU, int nK,
                                 float* __restrict__ out)
{
    int i = blockIdx.x * blockDim.x + threadIdx.x;
    if (i >= nU) return;
    float px = ux[3 * i], py = ux[3 * i + 1], pz = ux[3 * i + 2];
    float d0 = 1e30f, d1 = 1e30f, d2 = 1e30f;
    int i0 = 0, i1 = 0, i2 = 0;
    for (int j = 0; j < nK; ++j) {
        float dx = kx[3 * j] - px, dy = kx[3 * j + 1] - py, dz = kx[3 * j + 2] - pz;
        float d = dx * dx + dy * dy + dz * dz;
        if (d < d0)      { d2 = d1; i2 = i1; d1 = d0; i1 = i0; d0 = d; i0 = j; }
        else if (d < d1) { d2 = d1; i2 = i1; d1 = d; i1 = j; }
        else if (d < d2) { d2 = d; i2 = j; }
    }
    float w0 = 1.f / (d0 + 1e-8f), w1 = 1.f / (d1 + 1e-8f), w2 = 1.f / (d2 + 1e-8f);
    float s = w0 + w1 + w2; w0 /= s; w1 /= s; w2 /= s;
    for (int c = 0; c < C; ++c) {
        const float* r = kf + (size_t)c * nK;
        out[(size_t)c * nU + i] = w0 * r[i0] + w1 * r[i1] + w2 * r[i2];
    }
}

// ---------------------------------------------------------------------------
// Kron-style deconvolution: y[o, h*k+i, w*k+j] = sum_c x[c,h,w] * w[c,o,i,j] + b
// ---------------------------------------------------------------------------
__global__ void deconv_kernel(const float* __restrict__ x, const float* __restrict__ w,
                              const float* __restrict__ bias, int C, int O,
                              int H, int W, int k, float* __restrict__ out)
{
    int OH = H * k, OW = W * k;
    long idx = (long)blockIdx.x * blockDim.x + threadIdx.x;
    if (idx >= (long)O * OH * OW) return;
    int o = (int)(idx / ((long)OH * OW));
    int rem = (int)(idx % ((long)OH * OW));
    int oh = rem / OW, ow = rem % OW;
    int h = oh / k, ii = oh % k, ww = ow / k, jj = ow % k;
    float acc = bias[o];
    for (int c = 0; c < C; ++c)
        acc += x[((size_t)c * H + h) * W + ww] * w[(((size_t)c * O + o) * k + ii) * k + jj];
    out[idx] = acc;
}

// ---------------------------------------------------------------------------
// Host orchestration
// ---------------------------------------------------------------------------
struct AttP {
    const float *iacw, *iacb, *iag, *iab;
    const float *f1w, *f1b, *f2w, *f2b, *f3w, *f3b;
    const float *cw, *cb, *bg, *bb;
};

static const int   H_NPTS[5]      = {16384, 4096, 1024, 256, 64};
static const float H_RADIUS[4][2] = {{0.1f,0.5f},{0.5f,1.0f},{1.0f,2.0f},{2.0f,4.0f}};
static const int   H_NS[2]        = {16, 32};
static const int   H_MLPS[4][2][3]= {{{16,16,32},{32,32,64}},{{64,64,128},{64,96,128}},
                                     {{128,196,256},{128,196,256}},{{256,256,512},{256,384,512}}};
static const int   H_FPM[4][2]    = {{128,128},{256,256},{512,512},{512,512}};
static const int   H_IMGCH[5]     = {3, 64, 128, 256, 512};
static const int   H_PTCH[4]      = {96, 256, 512, 1024};
static const int   H_DECK[4]      = {2, 4, 8, 16};
static const int   H_CIN[4]       = {0, 96, 256, 512};

extern "C" void kernel_launch(void* const* d_in, const int* in_sizes, int n_in,
                              void* d_out, int out_size, void* d_ws, size_t ws_size,
                              hipStream_t stream)
{
    (void)in_sizes; (void)n_in; (void)out_size; (void)ws_size;
    const float* pcloud = (const float*)d_in[0];      // [16384,3]
    const float* image  = (const float*)d_in[1];      // [3,384,1280]
    const float* xy     = (const float*)d_in[2];      // [16384,2]

    // ---- parameter pointers (setup_inputs insertion order, recursively) ----
    int cursor = 3;
    auto nextp = [&]() -> const float* { return (const float*)d_in[cursor++]; };

    const float *sa_w[4][2][3], *sa_g[4][2][3], *sa_b[4][2][3];
    for (int k = 0; k < 4; ++k)
        for (int j = 0; j < 2; ++j)
            for (int l = 0; l < 3; ++l) { sa_w[k][j][l]=nextp(); sa_g[k][j][l]=nextp(); sa_b[k][j][l]=nextp(); }
    struct { const float *w1, *g, *b, *w2; } imgp[4];
    for (int i = 0; i < 4; ++i) { imgp[i].w1=nextp(); imgp[i].g=nextp(); imgp[i].b=nextp(); imgp[i].w2=nextp(); }
    auto readAtt = [&](AttP& a) {
        a.iacw=nextp(); a.iacb=nextp(); a.iag=nextp(); a.iab=nextp();
        a.f1w=nextp(); a.f1b=nextp(); a.f2w=nextp(); a.f2b=nextp();
        a.f3w=nextp(); a.f3b=nextp(); a.cw=nextp(); a.cb=nextp(); a.bg=nextp(); a.bb=nextp();
    };
    AttP fusp[4], finp;
    for (int i = 0; i < 4; ++i) readAtt(fusp[i]);
    const float *dec_w[4], *dec_b[4];
    for (int i = 0; i < 4; ++i) { dec_w[i]=nextp(); dec_b[i]=nextp(); }
    const float *ifw=nextp(), *ifb=nextp(), *ifg=nextp(), *ifbeta=nextp();
    readAtt(finp);
    const float *fp_w[4][2], *fp_g[4][2], *fp_b[4][2];
    for (int k = 0; k < 4; ++k)
        for (int l = 0; l < 2; ++l) { fp_w[k][l]=nextp(); fp_g[k][l]=nextp(); fp_b[k][l]=nextp(); }

    // ---- workspace bump allocator ----
    size_t off = 0;
    auto AF = [&](size_t n) -> float* {
        float* p = (float*)((char*)d_ws + off);
        off += ((n * sizeof(float)) + 255ull) / 256ull * 256ull;
        return p;
    };
    auto AI = [&](size_t n) -> int* {
        int* p = (int*)((char*)d_ws + off);
        off += ((n * sizeof(int)) + 255ull) / 256ull * 256ull;
        return p;
    };

    const size_t HW0 = (size_t)384 * 1280;
    float* XN   = AF(16384 * 2);
    float* NX[4]; for (int k = 0; k < 4; ++k) NX[k] = AF((size_t)H_NPTS[k + 1] * 3);
    float* LXY[5]; LXY[0] = XN;
    for (int k = 1; k < 5; ++k) LXY[k] = AF((size_t)H_NPTS[k] * 2);
    float* LF[5];
    LF[0] = AF((size_t)128 * 16384);
    LF[1] = AF((size_t)256 * 4096);
    LF[2] = AF((size_t)512 * 1024);
    LF[3] = AF((size_t)512 * 256);
    LF[4] = AF((size_t)1024 * 64);
    float* IMGB[5]; IMGB[0] = nullptr;
    for (int k = 1; k < 5; ++k)
        IMGB[k] = AF((size_t)H_IMGCH[k] * (384 >> k) * (1280 >> k));
    float* CONV1 = AF((size_t)64 * HW0);
    float* DEC   = AF((size_t)64 * HW0);
    float* IMF   = AF((size_t)32 * HW0);
    float* PING  = AF((size_t)6 << 20);
    float* PONG  = AF((size_t)6 << 20);
    float* XCAT  = AF((size_t)6 << 20);
    float* LIF   = AF((size_t)1 << 20);
    float* IMGG  = AF((size_t)1 << 20);
    float* IG    = AF((size_t)32 * 16384);
    float* RI    = AF((size_t)1 << 20);
    float* RP    = AF((size_t)1 << 20);
    float* ATT   = AF(16384);
    float* IMGNEW= AF((size_t)1 << 22);
    float* OUTF  = AF((size_t)128 * 16384);
    float* MV    = AF(2048);
    float* FPSD  = AF(16384);
    int*   FPSI  = AI(4096);
    int*   GIDX  = AI((size_t)4096 * 32);

    auto cdiv = [](long a, long b) -> int { return (int)((a + b - 1) / b); };

    auto gemm = [&](const float* W, const float* X, const float* bias, float* Y,
                    int O, int K, int N, int mode, int Hin, int Win,
                    int Hout, int Wout, int stride) {
        dim3 grid(cdiv(N, 128), cdiv(O, 32));
        wmma_gemm_kernel<<<grid, 128, 0, stream>>>(W, X, bias, Y, O, K, N, mode,
                                                   Hin, Win, Hout, Wout, stride);
    };
    auto bnrelu = [&](float* Y, const float* g, const float* b, int O, int N, int relu) {
        stats_kernel<<<O, 256, 0, stream>>>(Y, N, MV);
        long tot = (long)O * N;
        bn_relu_kernel<<<cdiv(tot, 256), 256, 0, stream>>>(Y, MV, g, b, O, N, relu);
    };
    auto fusion = [&](const AttP& p, const float* point, const float* img,
                      int pcch, int icch, int outch, int n, float* outbuf) {
        int rc = pcch / 4;
        gemm(p.f1w, img,   p.f1b, RI, rc, icch, n, 0, 0, 0, 0, 0, 1);
        gemm(p.f2w, point, p.f2b, RP, rc, pcch, n, 0, 0, 0, 0, 0, 1);
        add_tanh_kernel<<<cdiv((long)rc * n, 256), 256, 0, stream>>>(RI, RP, rc * n);
        gemm(p.f3w, RI, p.f3b, ATT, 1, rc, n, 0, 0, 0, 0, 0, 1);
        sigmoid_kernel<<<cdiv(n, 256), 256, 0, stream>>>(ATT, n);
        gemm(p.iacw, img, p.iacb, IMGNEW, pcch, icch, n, 0, 0, 0, 0, 0, 1);
        bnrelu(IMGNEW, p.iag, p.iab, pcch, n, 1);
        mul_att_kernel<<<cdiv((long)pcch * n, 256), 256, 0, stream>>>(IMGNEW, ATT, pcch, n);
        hipMemcpyAsync(XCAT, point, (size_t)pcch * n * sizeof(float),
                       hipMemcpyDeviceToDevice, stream);
        hipMemcpyAsync(XCAT + (size_t)pcch * n, IMGNEW, (size_t)pcch * n * sizeof(float),
                       hipMemcpyDeviceToDevice, stream);
        gemm(p.cw, XCAT, p.cb, outbuf, outch, 2 * pcch, n, 0, 0, 0, 0, 0, 1);
        bnrelu(outbuf, p.bg, p.bb, outch, n, 1);
    };

    // ---- l_xy[0] = normalized coords ----
    xn_kernel<<<cdiv(16384, 256), 256, 0, stream>>>(xy, XN, 16384);

    const float* lxyz[5]; lxyz[0] = pcloud;
    int chan[5] = {0, 96, 256, 512, 1024};

    // =========================== SA encoder levels ==========================
    for (int k = 0; k < 4; ++k) {
        int Nk = H_NPTS[k], np = H_NPTS[k + 1];
        fps_kernel<<<1, 1024, 0, stream>>>(lxyz[k], Nk, np, FPSI, FPSD);
        gather3_kernel<<<cdiv(np, 256), 256, 0, stream>>>(lxyz[k], FPSI, NX[k], np);
        gather2_kernel<<<cdiv(np, 256), 256, 0, stream>>>(LXY[k], FPSI, LXY[k + 1], np);

        int coff = 0;
        for (int j = 0; j < 2; ++j) {
            int ns = H_NS[j];
            float r = H_RADIUS[k][j];
            ballq_kernel<<<cdiv(np, 128), 128, 0, stream>>>(NX[k], lxyz[k], np, Nk,
                                                            r * r, ns, GIDX);
            int NN = np * ns;
            const float* fk = (k == 0) ? nullptr : LF[k];
            build_group_kernel<<<cdiv(NN, 256), 256, 0, stream>>>(
                lxyz[k], fk, H_CIN[k], Nk, GIDX, NX[k], np, ns, PING);
            float* cur = PING; float* nxt = PONG;
            int cin = H_CIN[k] + 3;
            for (int l = 0; l < 3; ++l) {
                int co = H_MLPS[k][j][l];
                gemm(sa_w[k][j][l], cur, nullptr, nxt, co, cin, NN, 0, 0, 0, 0, 0, 1);
                bnrelu(nxt, sa_g[k][j][l], sa_b[k][j][l], co, NN, 1);
                float* t = cur; cur = nxt; nxt = t; cin = co;
            }
            rowmax_kernel<<<cdiv((long)cin * np, 256), 256, 0, stream>>>(
                cur, cin, np, ns, LIF + (size_t)coff * np);
            coff += cin;
        }

        // image basic block: conv3x3(s1)+BN+ReLU, then conv3x3(s2)
        int Hk = 384 >> k, Wk = 1280 >> k, Ho = Hk >> 1, Wo = Wk >> 1;
        const float* imin = (k == 0) ? image : IMGB[k];
        int Ci = H_IMGCH[k], Co = H_IMGCH[k + 1];
        gemm(imgp[k].w1, imin, nullptr, CONV1, Co, Ci * 9, Hk * Wk, 1, Hk, Wk, Hk, Wk, 1);
        bnrelu(CONV1, imgp[k].g, imgp[k].b, Co, Hk * Wk, 1);
        gemm(imgp[k].w2, CONV1, nullptr, IMGB[k + 1], Co, Co * 9, Ho * Wo, 1, Hk, Wk, Ho, Wo, 2);

        grid_sample_kernel<<<cdiv((long)Co * np, 256), 256, 0, stream>>>(
            IMGB[k + 1], Co, Ho, Wo, LXY[k + 1], np, IMGG);
        fusion(fusp[k], LIF, IMGG, H_PTCH[k], Co, H_PTCH[k], np, LF[k + 1]);
        chan[k + 1] = H_PTCH[k];
        lxyz[k + 1] = NX[k];
    }

    // =========================== FP decoder =================================
    for (int s = 0; s < 4; ++s) {
        int ui = 3 - s;
        int nU = H_NPTS[ui], nKn = H_NPTS[ui + 1];
        int Ck = chan[ui + 1], Cu = chan[ui];
        fp_interp_kernel<<<cdiv(nU, 128), 128, 0, stream>>>(
            lxyz[ui], lxyz[ui + 1], LF[ui + 1], Ck, nU, nKn, XCAT);
        if (Cu > 0)
            hipMemcpyAsync(XCAT + (size_t)Ck * nU, LF[ui],
                           (size_t)Cu * nU * sizeof(float),
                           hipMemcpyDeviceToDevice, stream);
        int cin = Ck + Cu;
        float* cur = XCAT;
        float* bufs[2] = {PING, PONG};
        for (int l = 0; l < 2; ++l) {
            int co = H_FPM[ui][l];
            gemm(fp_w[ui][l], cur, nullptr, bufs[l], co, cin, nU, 0, 0, 0, 0, 0, 1);
            bnrelu(bufs[l], fp_g[ui][l], fp_b[ui][l], co, nU, 1);
            cur = bufs[l]; cin = co;
        }
        hipMemcpyAsync(LF[ui], cur, (size_t)cin * nU * sizeof(float),
                       hipMemcpyDeviceToDevice, stream);
        chan[ui] = cin;
    }

    // ================= deconv pyramid + 1x1 fuse + final fusion =============
    for (int i = 0; i < 4; ++i) {
        int Hi = 384 >> (i + 1), Wi = 1280 >> (i + 1), kk = H_DECK[i];
        long tot = (long)16 * HW0;
        deconv_kernel<<<cdiv(tot, 256), 256, 0, stream>>>(
            IMGB[i + 1], dec_w[i], dec_b[i], H_IMGCH[i + 1], 16, Hi, Wi, kk,
            DEC + (size_t)i * 16 * HW0);
    }
    gemm(ifw, DEC, ifb, IMF, 32, 64, (int)HW0, 0, 0, 0, 0, 0, 1);
    bnrelu(IMF, ifg, ifbeta, 32, (int)HW0, 1);
    grid_sample_kernel<<<cdiv((long)32 * 16384, 256), 256, 0, stream>>>(
        IMF, 32, 384, 1280, XN, 16384, IG);
    fusion(finp, LF[0], IG, 128, 32, 128, 16384, OUTF);

    // ---- outputs: (l_xyz[0], out_feat) concatenated flat ----
    float* out = (float*)d_out;
    hipMemcpyAsync(out, pcloud, (size_t)16384 * 3 * sizeof(float),
                   hipMemcpyDeviceToDevice, stream);
    hipMemcpyAsync(out + 16384 * 3, OUTF, (size_t)128 * 16384 * sizeof(float),
                   hipMemcpyDeviceToDevice, stream);
}
